// HyperGraphLayerSparse_19344532701487
// MI455X (gfx1250) — compile-verified, matrix operand order checked by reading, not run on hardware
//
#include <hip/hip_runtime.h>
#include <hip/hip_bf16.h>
#include <math.h>

#define ALPHA 0.2f
#define N1 4096
#define N2 8192
#define DD 256

typedef __bf16 bf16_t;
typedef bf16_t v16bf __attribute__((ext_vector_type(16)));
typedef bf16_t v8bf  __attribute__((ext_vector_type(8)));
typedef float  v8f   __attribute__((ext_vector_type(8)));

// ---------------------------------------------------------------------------
// WMMA 16-bit fragment layout (ISA 7.12.2): for lane-half h, vector elements
// 0..7 hold K = h*8 + 0..7 and elements 8..15 hold K = 16 + h*8 + 0..7.
// Both groups are contiguous in K -> two 16B loads when K is the fast axis.
// ---------------------------------------------------------------------------
__device__ __forceinline__ v16bf frag_from(const bf16_t* p) {
  v8bf lo = *(const v8bf*)(p);
  v8bf hi = *(const v8bf*)(p + 16);
  return __builtin_shufflevector(lo, hi, 0, 1, 2, 3, 4, 5, 6, 7,
                                         8, 9, 10, 11, 12, 13, 14, 15);
}

// A fragment (M=16, K=32) from row-major LDS tile [16][32]; lane%16 = M row.
__device__ __forceinline__ v16bf load_A(const bf16_t* sA, int lane) {
  return frag_from(sA + (lane & 15) * 32 + (lane >> 4) * 8);
}

// B fragment (K=32, N=16) from TRANSPOSED global [n][ldk]; lane%16 = n column.
__device__ __forceinline__ v16bf load_Bt(const bf16_t* Bt, int ldk, int k0, int n, int lane) {
  return frag_from(Bt + (size_t)n * ldk + k0 + (lane >> 4) * 8);
}

#define WMMA_BF16(A, B, C) \
  __builtin_amdgcn_wmma_f32_16x16x32_bf16(false, (A), false, (B), (short)0, (C), false, false)

// W[256][256] f32 row-major -> Wt[n][k] bf16 (B operand layout)
__global__ void transpose_w(const float* __restrict__ W, bf16_t* __restrict__ Wt) {
  int n = blockIdx.x;
  int k = threadIdx.x;
  Wt[n * DD + k] = (bf16_t)W[k * DD + n];
}

// ---------------------------------------------------------------------------
// C[M x 256] = A[M x 256] @ B[256 x 256]; B given transposed bf16 [n][256].
// Block = 16 rows x 256 cols, 8 waves, each wave owns two 16x16 D tiles.
// Writes f32 row-major and bf16 TRANSPOSED [n][mtot] copies.
// ---------------------------------------------------------------------------
template <bool A_IS_BF16>
__global__ void gemm_k256(const void* Aptr, const bf16_t* __restrict__ Bt,
                          float* __restrict__ outF, bf16_t* __restrict__ outT, int mtot) {
  __shared__ bf16_t sA[16 * 32];
  const int tid = threadIdx.x, lane = tid & 31, wave = tid >> 5;
  const int half = lane >> 4;
  const int row0 = blockIdx.x * 16;
  const int n_base = wave * 32;
  v8f acc0 = {}, acc1 = {};
  for (int k0 = 0; k0 < 256; k0 += 32) {
#pragma unroll
    for (int q = 0; q < 2; ++q) {
      int idx = tid + q * 256;            // 512 elements, 2 per thread
      int r = idx >> 5, c = idx & 31;
      float v;
      if (A_IS_BF16) v = (float)((const bf16_t*)Aptr)[(row0 + r) * DD + k0 + c];
      else           v = ((const float*)Aptr)[(row0 + r) * DD + k0 + c];
      sA[idx] = (bf16_t)v;
    }
    __syncthreads();
    v16bf a  = load_A(sA, lane);
    v16bf b0 = load_Bt(Bt, DD, k0, n_base + (lane & 15), lane);
    v16bf b1 = load_Bt(Bt, DD, k0, n_base + 16 + (lane & 15), lane);
    acc0 = WMMA_BF16(a, b0, acc0);
    acc1 = WMMA_BF16(a, b1, acc1);
    __syncthreads();
  }
  const int n0 = n_base + (lane & 15);
  v8bf t0, t1;
#pragma unroll
  for (int v = 0; v < 8; ++v) {
    int row = row0 + half * 8 + v;        // accumulator rows are consecutive in M
    outF[row * DD + n0]      = acc0[v];
    outF[row * DD + n0 + 16] = acc1[v];
    t0[v] = (bf16_t)acc0[v];
    t1[v] = (bf16_t)acc1[v];
  }
  *(v8bf*)(outT + (size_t)n0 * mtot + row0 + half * 8) = t0;
  *(v8bf*)(outT + (size_t)(n0 + 16) * mtot + row0 + half * 8) = t1;
}

// ---------------------------------------------------------------------------
// edge = (adj @ x4att) / degree; degree = rowsum(adj) fused into A staging so
// adj is read exactly once. B = x4att transposed bf16 [n][N2] (L2 resident).
// Output: edge row-major bf16 (A operand of the next GEMM).
// ---------------------------------------------------------------------------
__global__ void edge_gemm(const float* __restrict__ adj, const bf16_t* __restrict__ x4t,
                          bf16_t* __restrict__ edgeB) {
  __shared__ bf16_t sA[16 * 32];
  __shared__ float sdeg[16];
  const int tid = threadIdx.x, lane = tid & 31, wave = tid >> 5;
  const int half = lane >> 4;
  const int row0 = blockIdx.x * 16;       // edge rows i
  const int n_base = wave * 32;
  if (tid < 16) sdeg[tid] = 0.f;
  float dp0 = 0.f, dp1 = 0.f;             // degree partials (rows tid>>5, 8+(tid>>5))
  v8f acc0 = {}, acc1 = {};
  for (int k0 = 0; k0 < N2; k0 += 32) {
    {
      int r = tid >> 5, c = tid & 31;
      float v = adj[(row0 + r) * N2 + k0 + c];
      sA[tid] = (bf16_t)v;  dp0 += v;
      int idx = tid + 256;
      r = idx >> 5; c = idx & 31;
      v = adj[(row0 + r) * N2 + k0 + c];
      sA[idx] = (bf16_t)v;  dp1 += v;
    }
    __syncthreads();
    v16bf a  = load_A(sA, lane);
    v16bf b0 = load_Bt(x4t, N2, k0, n_base + (lane & 15), lane);
    v16bf b1 = load_Bt(x4t, N2, k0, n_base + 16 + (lane & 15), lane);
    acc0 = WMMA_BF16(a, b0, acc0);
    acc1 = WMMA_BF16(a, b1, acc1);
    __syncthreads();
  }
  atomicAdd(&sdeg[tid >> 5], dp0);        // ds_add_f32; sums of 0/1 are exact
  atomicAdd(&sdeg[8 + (tid >> 5)], dp1);
  __syncthreads();
  const int n0 = n_base + (lane & 15);
#pragma unroll
  for (int v = 0; v < 8; ++v) {
    int m = half * 8 + v;
    float inv = 1.f / sdeg[m];
    edgeB[(row0 + m) * DD + n0]      = (bf16_t)(acc0[v] * inv);
    edgeB[(row0 + m) * DD + n0 + 16] = (bf16_t)(acc1[v] * inv);
  }
}

// out[r] = Mtx[r,:] . vec  (one wave per row)
__global__ void rowdot(const float* __restrict__ Mtx, const float* __restrict__ vec,
                       float* __restrict__ out, int nrows) {
  int gwave = (blockIdx.x * blockDim.x + threadIdx.x) >> 5;
  int lane = threadIdx.x & 31;
  if (gwave >= nrows) return;
  float s = 0.f;
  for (int n = lane; n < DD; n += 32) s += Mtx[gwave * DD + n] * vec[n];
#pragma unroll
  for (int off = 16; off > 0; off >>= 1) s += __shfl_down(s, off, 32);
  if (lane == 0) out[gwave] = s;
}

// Online (max, sum-exp) over a 256-long chunk of the edge axis, per node j.
__global__ void softmax_partial(const float* __restrict__ adj, const float* __restrict__ se,
                                const float* __restrict__ sn, float* __restrict__ mpart,
                                float* __restrict__ lpart) {
  const int j = blockIdx.x * blockDim.x + threadIdx.x;
  const int chunk = blockIdx.y;           // 16 chunks of N1/16 = 256 edges
  const float snj = sn[j];
  float m = -__builtin_inff(), l = 0.f;
  const int i0 = chunk * (N1 / 16);
  for (int i = i0; i < i0 + N1 / 16; ++i) {
    float av = adj[i * N2 + j];           // coalesced across j
    float sc = se[i] + snj;
    float val = (av > 0.f) ? (sc > 0.f ? sc : ALPHA * sc) : -9.0e15f;
    float mn = fmaxf(m, val);
    l = l * __expf(m - mn) + __expf(val - mn);
    m = mn;
  }
  mpart[chunk * N2 + j] = m;
  lpart[chunk * N2 + j] = l;
}

__global__ void softmax_reduce(const float* __restrict__ mpart, const float* __restrict__ lpart,
                               float* __restrict__ mj, float* __restrict__ linv) {
  const int j = blockIdx.x * blockDim.x + threadIdx.x;
  float m = -__builtin_inff();
#pragma unroll
  for (int c = 0; c < 16; ++c) m = fmaxf(m, mpart[c * N2 + j]);
  float l = 0.f;
#pragma unroll
  for (int c = 0; c < 16; ++c) l += lpart[c * N2 + j] * __expf(mpart[c * N2 + j] - m);
  mj[j] = m;
  linv[j] = 1.f / l;                      // l >= 1 always (max term contributes exp(0))
}

// ---------------------------------------------------------------------------
// out[j,:] = leaky_relu( (1/l_j) * sum_i exp(attn(i,j)-m_j) * edge_4att[i,:] )
// P tile (16 j x 32 i) recomputed in LDS as bf16; B = edge_4att^T bf16 [n][N1].
// ---------------------------------------------------------------------------
__global__ void attn_out(const float* __restrict__ adj, const float* __restrict__ se,
                         const float* __restrict__ sn, const float* __restrict__ mj,
                         const float* __restrict__ linv, const bf16_t* __restrict__ e4t,
                         float* __restrict__ out) {
  __shared__ bf16_t sP[16 * 32];
  __shared__ float s_m[16], s_li[16], s_sn[16];
  const int tid = threadIdx.x, lane = tid & 31, wave = tid >> 5;
  const int half = lane >> 4;
  const int j0 = blockIdx.x * 16;
  const int n_base = wave * 32;
  if (tid < 16) { s_m[tid] = mj[j0 + tid]; s_li[tid] = linv[j0 + tid]; s_sn[tid] = sn[j0 + tid]; }
  __syncthreads();
  v8f acc0 = {}, acc1 = {};
  for (int i0 = 0; i0 < N1; i0 += 32) {
#pragma unroll
    for (int q = 0; q < 2; ++q) {
      int idx = tid + q * 256;            // 32 i x 16 j
      int ioff = idx >> 4, jj = idx & 15;
      float av = adj[(i0 + ioff) * N2 + j0 + jj];
      float sc = se[i0 + ioff] + s_sn[jj];
      float val = (av > 0.f) ? (sc > 0.f ? sc : ALPHA * sc) : -9.0e15f;
      sP[jj * 32 + ioff] = (bf16_t)__expf(val - s_m[jj]);  // in [0,1]
    }
    __syncthreads();
    v16bf a  = load_A(sP, lane);
    v16bf b0 = load_Bt(e4t, N1, i0, n_base + (lane & 15), lane);
    v16bf b1 = load_Bt(e4t, N1, i0, n_base + 16 + (lane & 15), lane);
    acc0 = WMMA_BF16(a, b0, acc0);
    acc1 = WMMA_BF16(a, b1, acc1);
    __syncthreads();
  }
  const int n0 = n_base + (lane & 15);
#pragma unroll
  for (int v = 0; v < 8; ++v) {
    int m = half * 8 + v;
    float li = s_li[m];
    float x0 = acc0[v] * li, x1 = acc1[v] * li;
    x0 = x0 > 0.f ? x0 : ALPHA * x0;
    x1 = x1 > 0.f ? x1 : ALPHA * x1;
    out[(j0 + m) * DD + n0]      = x0;
    out[(j0 + m) * DD + n0 + 16] = x1;
  }
}

extern "C" void kernel_launch(void* const* d_in, const int* in_sizes, int n_in,
                              void* d_out, int out_size, void* d_ws, size_t ws_size,
                              hipStream_t stream) {
  const float* x    = (const float*)d_in[0];   // (N2, 256)
  const float* adj  = (const float*)d_in[1];   // (N1, N2)
  const float* wv2e = (const float*)d_in[2];   // (256, 256)
  const float* we2v = (const float*)d_in[3];   // (256, 256)
  const float* a    = (const float*)d_in[4];   // (1, 512)
  float* out = (float*)d_out;                  // (N2, 256)

  char* ws = (char*)d_ws;
  size_t off = 0;
  auto alloc = [&](size_t bytes) -> void* {
    void* p = ws + off;
    off += (bytes + 255) & ~(size_t)255;
    return p;
  };
  float*  x4f    = (float*) alloc((size_t)N2 * DD * 4);  // x_4att f32 row-major
  bf16_t* x4t    = (bf16_t*)alloc((size_t)DD * N2 * 2);  // x_4att^T bf16 [n][N2]
  float*  e4f    = (float*) alloc((size_t)N1 * DD * 4);  // edge_4att f32 row-major
  bf16_t* e4t    = (bf16_t*)alloc((size_t)DD * N1 * 2);  // edge_4att^T bf16 [n][N1]
  bf16_t* edgeB  = (bf16_t*)alloc((size_t)N1 * DD * 2);  // edge bf16 row-major
  bf16_t* wv2e_t = (bf16_t*)alloc((size_t)DD * DD * 2);
  bf16_t* we2v_t = (bf16_t*)alloc((size_t)DD * DD * 2);
  float*  sn     = (float*) alloc((size_t)N2 * 4);
  float*  se     = (float*) alloc((size_t)N1 * 4);
  float*  mpart  = (float*) alloc((size_t)16 * N2 * 4);
  float*  lpart  = (float*) alloc((size_t)16 * N2 * 4);
  float*  mj     = (float*) alloc((size_t)N2 * 4);
  float*  linv   = (float*) alloc((size_t)N2 * 4);

  transpose_w<<<DD, DD, 0, stream>>>(wv2e, wv2e_t);
  transpose_w<<<DD, DD, 0, stream>>>(we2v, we2v_t);
  gemm_k256<false><<<N2 / 16, 256, 0, stream>>>((const void*)x, wv2e_t, x4f, x4t, N2);
  rowdot<<<N2 / 8, 256, 0, stream>>>(x4f, a, sn, N2);
  edge_gemm<<<N1 / 16, 256, 0, stream>>>(adj, x4t, edgeB);
  gemm_k256<true><<<N1 / 16, 256, 0, stream>>>((const void*)edgeB, we2v_t, e4f, e4t, N1);
  rowdot<<<N1 / 8, 256, 0, stream>>>(e4f, a + DD, se, N1);
  softmax_partial<<<dim3(N2 / 256, 16), 256, 0, stream>>>(adj, se, sn, mpart, lpart);
  softmax_reduce<<<N2 / 256, 256, 0, stream>>>(mpart, lpart, mj, linv);
  attn_out<<<N2 / 16, 256, 0, stream>>>(adj, se, sn, mj, linv, e4t, out);
}
